// RouteExactNgramMemory_1717986918577
// MI455X (gfx1250) — matrix operand order repaired and checked by reading, not compiled, not run on hardware
//
#include <hip/hip_runtime.h>
#include <hip/hip_bf16.h>
#include <stdint.h>

// ---------------------------------------------------------------------------
// RouteExactNgramMemory forward for MI455X (gfx1250, wave32).
//
// Stage 1 (WMMA): sign-bit quantize q_logits -> 4-bit codes, expressed as an
//   exact f16 matmul  codes[t,r] = sum_h bit(q[t,h]) * W[h,r],
//   W[h,r] = 2^(h&3) iff h>>2==r (block-diagonal), done per 16x16 tile with
//   two v_wmma_f32_16x16x32_f16 (K=64 hidden per route tile). All values are
//   small integers -> exact in f16/f32.
// Stage 2: n-gram hash gather. 4 lanes per 64-byte table row => coalesced
//   global_load_b128; output streamed with non-temporal b128 stores so the
//   64MB output does not evict table_3 (64MB) from the 192MB L2 — the gathers
//   then hit L2 instead of HBM.  HBM floor ~140MB => ~6us at 23.3 TB/s.
// ---------------------------------------------------------------------------

typedef __attribute__((ext_vector_type(16))) _Float16 v16h;
typedef __attribute__((ext_vector_type(8)))  float    v8f;
typedef __attribute__((ext_vector_type(4)))  float    f32x4;

#define BSZ    2
#define TLEN   1024
#define HIDDEN 1024
#define RIDX   256          // num routes
#define MEMD   16           // mem_dim
#define NTOK   (BSZ * TLEN) // 2048 tokens

// ---- Kernel 1: WMMA quantization ------------------------------------------
// One wave per (16-token x 16-route) tile. 2048 tiles = 2048 waves.
__global__ __launch_bounds__(256)
void RouteNgram_quantize_wmma(const float* __restrict__ q,
                              uint8_t* __restrict__ codes)
{
    const int lane = threadIdx.x & 31;
    const int wave = threadIdx.x >> 5;
    const int gw   = blockIdx.x * 8 + wave;   // 0..2047
    const int mt   = gw >> 4;                 // token tile 0..127
    const int nt   = gw & 15;                 // route tile 0..15
    const int m    = lane & 15;               // M (token-in-tile) / N for B,D
    const int khi  = lane >> 4;               // lane-half selects K sub-range

    const int token = mt * 16 + m;            // 0..2047 flat (b,t)
    const float* row = q + (size_t)token * HIDDEN + nt * 64; // 64 hidden / tile

    v8f acc = {};
    #pragma unroll
    for (int s = 0; s < 2; ++s) {             // two K=32 steps cover K=64
        // A layout (16-bit A 16x32, ISA 7.12.2): lanes 0-15 hold K 0..7,16..23;
        // lanes 16-31 hold K 8..15,24..31 (M = lane&15).
        const float* p = row + s * 32 + khi * 8;
        f32x4 x0 = __builtin_nontemporal_load((const f32x4*)(p));
        f32x4 x1 = __builtin_nontemporal_load((const f32x4*)(p + 4));
        f32x4 x2 = __builtin_nontemporal_load((const f32x4*)(p + 16));
        f32x4 x3 = __builtin_nontemporal_load((const f32x4*)(p + 20));
        v16h a;
        #pragma unroll
        for (int e = 0; e < 4; ++e) {
            a[e]      = (x0[e] > 0.f) ? (_Float16)1.0f : (_Float16)0.0f;
            a[e + 4]  = (x1[e] > 0.f) ? (_Float16)1.0f : (_Float16)0.0f;
            a[e + 8]  = (x2[e] > 0.f) ? (_Float16)1.0f : (_Float16)0.0f;
            a[e + 12] = (x3[e] > 0.f) ? (_Float16)1.0f : (_Float16)0.0f;
        }
        // B layout (16-bit B 32x16): lanes 0-15 hold K 0..7,16..23;
        // lanes 16-31 hold K 8..15,24..31 (same as A); N = lane&15. Synthesize block-diag W.
        v16h bfrag;
        #pragma unroll
        for (int j = 0; j < 16; ++j) {
            const int k = (j < 8) ? (8 * khi + j) : (16 + 8 * khi + (j - 8));
            const bool nz = ((k >> 2) == (m - s * 8));
            bfrag[j] = nz ? (_Float16)(float)(1 << (k & 3)) : (_Float16)0.0f;
        }
        acc = __builtin_amdgcn_wmma_f32_16x16x32_f16(
                  false, a, false, bfrag, (short)0, acc, false, false);
    }

    // D layout (32-bit 16x16): VGPR p -> M = p + 8*khi, N = lane&15.
    #pragma unroll
    for (int p = 0; p < 8; ++p) {
        const int tok2 = mt * 16 + p + khi * 8;
        const int r    = nt * 16 + m;
        codes[(size_t)tok2 * RIDX + r] = (uint8_t)(int)(acc[p] + 0.5f);
    }
}

// ---- Kernel 2: n-gram hash gather ------------------------------------------
// 4 lanes cooperate on each 64-byte memory row (b128 per lane).
__global__ __launch_bounds__(256)
void RouteNgram_gather(const uint8_t* __restrict__ codes,
                       const float* __restrict__ t2,
                       const float* __restrict__ t3,
                       float* __restrict__ out)
{
    const unsigned gtid = blockIdx.x * 256u + threadIdx.x;
    const int d4 = gtid & 3;            // which float4 of the 16-float row
    const unsigned v = gtid >> 2;       // vector id over (b,t,o,r)
    const int r = v & 255;
    const int o = (v >> 8) & 1;
    const int t = (v >> 9) & 1023;
    const int b = (int)(v >> 19);

    const size_t cb = (size_t)(b * TLEN + t) * RIDX + r;
    f32x4 val = {0.f, 0.f, 0.f, 0.f};   // padded first n-1 steps are zero

    if (o == 0) {                       // bigram branch -> table_2
        if (t >= 1) {
            const int c0 = codes[cb];
            const int c1 = codes[cb - RIDX];
            const int gidx = c1 + (c0 << 4) + (r << 8);
            val = *(const f32x4*)(t2 + (size_t)gidx * MEMD + d4 * 4);
        }
    } else {                            // trigram branch -> table_3
        if (t >= 2) {
            const int c0 = codes[cb];
            const int c1 = codes[cb - RIDX];
            const int c2 = codes[cb - 2 * RIDX];
            const int gidx = c2 + (c1 << 4) + (c0 << 8) + (r << 12);
            val = *(const f32x4*)(t3 + (size_t)gidx * MEMD + d4 * 4);
        }
    }
    // out[b,t,o,r,d]; stream with NT store to keep tables resident in L2.
    const size_t ob = ((size_t)(b * TLEN + t) * 2 + o) * (RIDX * MEMD)
                    + (size_t)r * MEMD + d4 * 4;
    __builtin_nontemporal_store(val, (f32x4*)(out + ob));
}

// ---------------------------------------------------------------------------
extern "C" void kernel_launch(void* const* d_in, const int* in_sizes, int n_in,
                              void* d_out, int out_size, void* d_ws, size_t ws_size,
                              hipStream_t stream)
{
    (void)in_sizes; (void)n_in; (void)out_size; (void)ws_size;
    const float* q_logits = (const float*)d_in[0];
    const float* table_2  = (const float*)d_in[1];
    const float* table_3  = (const float*)d_in[2];
    float*       out      = (float*)d_out;
    uint8_t*     codes    = (uint8_t*)d_ws;   // needs B*T*R = 512 KB scratch

    // 2048 tiles, 8 waves (256 thr) per block -> 256 blocks, EXEC all-ones.
    RouteNgram_quantize_wmma<<<256, 256, 0, stream>>>(q_logits, codes);

    // B*T*O*R*4 lanes = 4,194,304 -> 16384 blocks of 256.
    RouteNgram_gather<<<(BSZ * TLEN * 2 * RIDX * 4) / 256, 256, 0, stream>>>(
        codes, table_2, table_3, out);
}
